// GIN_82111184765292
// MI455X (gfx1250) — matrix operand order repaired and checked by reading, not compiled, not run on hardware
//
#include <hip/hip_runtime.h>

// ---------------------------------------------------------------------------
// GIN (2x GINConv + sigmoid head MLP) for MI455X / gfx1250.
// Strategy:
//   - Edge aggregation is the HBM-bound part (~3.3 GB traffic): coalesced
//     float4 gathers + global f32 atomics.
//   - All 6 GEMMs ([100K x 128] x [128 x {128,64}]) run on v_wmma_f32_16x16x32_bf16
//     (bf16 A/B, f32 accumulate). Weights are pre-transposed+converted to bf16
//     once per launch so B-fragments are two contiguous 16B loads per lane (L2-resident).
//   - Layer-2 GEMMs and the whole head MLP are fused in one kernel; the
//     intermediate activations move between GEMM stages via per-wave LDS
//     staging tiles (transpose from C/D layout back to A layout).
// ---------------------------------------------------------------------------

typedef __bf16 bf16_t;
typedef __attribute__((ext_vector_type(16))) __bf16 v16bf;
typedef __attribute__((ext_vector_type(8)))  float  v8f;

#define LSTR 136   // LDS staging row stride in bf16 elements (128 + 8 pad)
#define KDIM 128

__device__ __forceinline__ bf16_t f2bf(float f) {
    // round-to-nearest-even f32 -> bf16
    unsigned u = __builtin_bit_cast(unsigned, f);
    u += 0x7fffu + ((u >> 16) & 1u);
    unsigned short h = (unsigned short)(u >> 16);
    return __builtin_bit_cast(bf16_t, h);
}

// ---------------- WMMA fragment builders (ISA 7.12.2 layouts) ----------------

// A fragment (16x32 bf16) computed on the fly from s*x[row] + agg[row].
// lane 0..15  : row = lane,    elems 0..7 = K kb+0..7,  elems 8..15 = K kb+16..23
// lane 16..31 : row = lane-16, elems 0..7 = K kb+8..15, elems 8..15 = K kb+24..31
__device__ __forceinline__ v16bf a_frag_global(const float* __restrict__ x,
                                               const float* __restrict__ agg,
                                               float s, long row, int kb, int lane) {
    int half8 = (lane >> 4) << 3;          // 0 or 8
    const float* px = x   + row * (long)KDIM + kb + half8;
    const float* pa = agg + row * (long)KDIM + kb + half8;
    v16bf f;
#pragma unroll
    for (int i = 0; i < 8; ++i) {
        f[i]     = f2bf(s * px[i]      + pa[i]);
        f[8 + i] = f2bf(s * px[16 + i] + pa[16 + i]);
    }
    return f;
}

// A fragment from a per-wave LDS staging tile (bf16, stride LSTR).
__device__ __forceinline__ v16bf a_frag_lds(const bf16_t* __restrict__ my,
                                            int kb, int lane) {
    const bf16_t* p = my + (lane & 15) * LSTR + kb + ((lane >> 4) << 3);
    v16bf f;
#pragma unroll
    for (int i = 0; i < 8; ++i) {
        f[i]     = p[i];
        f[8 + i] = p[16 + i];
    }
    return f;
}

// B fragment (32x16 bf16) from pre-transposed bf16 weights wT[N][128]:
// lane holds column n = nTile*16 + (lane&15), elems 0..7 = K kb+8h..+7,
// elems 8..15 = K kb+16+8h..+7 (h = lane>=16) -> two aligned 16-byte loads.
__device__ __forceinline__ v16bf b_frag(const bf16_t* __restrict__ wT,
                                        int nTile, int kb, int lane) {
    int n  = (nTile << 4) + (lane & 15);
    int ks = kb + ((lane >> 4) << 3);
    typedef __attribute__((ext_vector_type(8))) __bf16 v8bf;
    v8bf lo = *(const v8bf*)(wT + n * KDIM + ks);
    v8bf hi = *(const v8bf*)(wT + n * KDIM + ks + 16);
    return __builtin_shufflevector(lo, hi, 0,1,2,3,4,5,6,7,8,9,10,11,12,13,14,15);
}

__device__ __forceinline__ v8f wmma_bf16(v16bf a, v16bf b, v8f c) {
    return __builtin_amdgcn_wmma_f32_16x16x32_bf16(
        false, a, false, b, (short)0, c, false, false);
}

// GEMM over the LDS staging tile: acc[NT] tiles of 16x16, K = 128.
template <int NT>
__device__ __forceinline__ void gemm_from_lds(const bf16_t* __restrict__ my,
                                              const bf16_t* __restrict__ wT,
                                              v8f* acc, int lane) {
#pragma unroll
    for (int t = 0; t < NT; ++t)
#pragma unroll
        for (int j = 0; j < 8; ++j) acc[t][j] = 0.0f;
#pragma unroll
    for (int kb = 0; kb < KDIM; kb += 32) {
        v16bf a = a_frag_lds(my, kb, lane);
#pragma unroll
        for (int t = 0; t < NT; ++t)
            acc[t] = wmma_bf16(a, b_frag(wT, t, kb, lane), acc[t]);
    }
}

// Bias + ReLU the accumulators and write (transposing) into the LDS stage.
// C/D layout: lane holds col n = lane&15, rows v + 8*(lane>=16).
template <int NT>
__device__ __forceinline__ void stage_bias_relu(bf16_t* __restrict__ my,
                                                const v8f* acc,
                                                const float* __restrict__ bias,
                                                int lane) {
    int n  = lane & 15;
    int ro = (lane >> 4) << 3;
#pragma unroll
    for (int t = 0; t < NT; ++t) {
        int col  = (t << 4) + n;
        float bb = bias[col];
#pragma unroll
        for (int v = 0; v < 8; ++v) {
            float val = fmaxf(acc[t][v] + bb, 0.0f);
            my[(ro + v) * LSTR + col] = f2bf(val);
        }
    }
}

// ---------------------------- small utility kernels --------------------------

// Transpose + convert one f32 weight [K][Nc] -> bf16 [Nc][K].
__global__ void wprep_kernel(const float* __restrict__ w, bf16_t* __restrict__ wT,
                             int K, int Nc) {
    int i = blockIdx.x * blockDim.x + threadIdx.x;
    if (i >= K * Nc) return;
    int n = i / K;
    int k = i - n * K;
    wT[i] = f2bf(w[k * Nc + n]);
}

// agg[dst[e]] += feat[src[e]] : one thread per (edge, float4 chunk).
__global__ void scatter_kernel(const float* __restrict__ feat,
                               const int* __restrict__ src,
                               const int* __restrict__ dst,
                               float* __restrict__ agg, long nWork) {
    long i = (long)blockIdx.x * blockDim.x + threadIdx.x;
    if (i >= nWork) return;
    int e  = (int)(i >> 5);
    int c4 = (int)(i & 31) << 2;
    long s = src[e];
    long d = dst[e];
    const float4 v = *(const float4*)(feat + s * KDIM + c4);
    float* q = agg + d * KDIM + c4;
    atomicAdd(q + 0, v.x);
    atomicAdd(q + 1, v.y);
    atomicAdd(q + 2, v.z);
    atomicAdd(q + 3, v.w);
}

// --------------------------- fused GIN layer 1 -------------------------------
// h1 = relu( relu(((1+eps)x + agg) @ w1 + b1) @ w2 + b2 )
__global__ __launch_bounds__(256)
void gin_update_kernel(const float* __restrict__ x, const float* __restrict__ agg,
                       const float* __restrict__ epsP,
                       const bf16_t* __restrict__ w1T, const float* __restrict__ b1,
                       const bf16_t* __restrict__ w2T, const float* __restrict__ b2,
                       float* __restrict__ out, int N) {
    __shared__ bf16_t stage[8][16 * LSTR];
    const int lane = threadIdx.x & 31;
    const int wave = threadIdx.x >> 5;
    const long rowBase = (long)blockIdx.x * 128 + wave * 16;
    const float s = 1.0f + *epsP;
    bf16_t* my = &stage[wave][0];

    long r = rowBase + (lane & 15);
    if (r >= N) r = N - 1;   // clamp loads; stores are guarded

    // ---- GEMM1: (16x128) = A @ w1 ----
    v8f acc[8];
#pragma unroll
    for (int t = 0; t < 8; ++t)
#pragma unroll
        for (int j = 0; j < 8; ++j) acc[t][j] = 0.0f;
#pragma unroll
    for (int kb = 0; kb < KDIM; kb += 32) {
        v16bf a = a_frag_global(x, agg, s, r, kb, lane);
#pragma unroll
        for (int t = 0; t < 8; ++t)
            acc[t] = wmma_bf16(a, b_frag(w1T, t, kb, lane), acc[t]);
    }

    __syncthreads();
    stage_bias_relu<8>(my, acc, b1, lane);
    __syncthreads();

    // ---- GEMM2 + bias + outer ReLU -> global f32 ----
    v8f acc2[8];
    gemm_from_lds<8>(my, w2T, acc2, lane);

    int n  = lane & 15;
    int ro = (lane >> 4) << 3;
#pragma unroll
    for (int t = 0; t < 8; ++t) {
        int col  = (t << 4) + n;
        float bb = b2[col];
#pragma unroll
        for (int v = 0; v < 8; ++v) {
            long row = rowBase + ro + v;
            if (row < N)
                out[row * (long)KDIM + col] = fmaxf(acc2[t][v] + bb, 0.0f);
        }
    }
}

// --------------------- fused GIN layer 2 + head MLP --------------------------
// h2 = relu( relu(((1+eps)h1 + agg) @ w21 + b21) @ w22 + b22 )
// out = sigmoid( relu(h2 @ wm1 + bm1) @ wm2 + bm2 )      [N x 64]
__global__ __launch_bounds__(256)
void gin2_head_kernel(const float* __restrict__ h1, const float* __restrict__ agg,
                      const float* __restrict__ epsP,
                      const bf16_t* __restrict__ w21T, const float* __restrict__ b21,
                      const bf16_t* __restrict__ w22T, const float* __restrict__ b22,
                      const bf16_t* __restrict__ wm1T, const float* __restrict__ bm1,
                      const bf16_t* __restrict__ wm2T, const float* __restrict__ bm2,
                      float* __restrict__ out, int N) {
    __shared__ bf16_t stage[8][16 * LSTR];
    const int lane = threadIdx.x & 31;
    const int wave = threadIdx.x >> 5;
    const long rowBase = (long)blockIdx.x * 128 + wave * 16;
    const float s = 1.0f + *epsP;
    bf16_t* my = &stage[wave][0];

    long r = rowBase + (lane & 15);
    if (r >= N) r = N - 1;

    v8f acc[8];

    // ---- GEMM1: ((1+eps)h1 + agg) @ w21 ----
#pragma unroll
    for (int t = 0; t < 8; ++t)
#pragma unroll
        for (int j = 0; j < 8; ++j) acc[t][j] = 0.0f;
#pragma unroll
    for (int kb = 0; kb < KDIM; kb += 32) {
        v16bf a = a_frag_global(h1, agg, s, r, kb, lane);
#pragma unroll
        for (int t = 0; t < 8; ++t)
            acc[t] = wmma_bf16(a, b_frag(w21T, t, kb, lane), acc[t]);
    }
    __syncthreads();
    stage_bias_relu<8>(my, acc, b21, lane);
    __syncthreads();

    // ---- GEMM2: @ w22, ReLU (outer relu of layer 2) ----
    gemm_from_lds<8>(my, w22T, acc, lane);
    __syncthreads();
    stage_bias_relu<8>(my, acc, b22, lane);
    __syncthreads();

    // ---- GEMM3 (head hidden): @ wm1, ReLU ----
    gemm_from_lds<8>(my, wm1T, acc, lane);
    __syncthreads();
    stage_bias_relu<8>(my, acc, bm1, lane);
    __syncthreads();

    // ---- GEMM4 (head out, N=64): @ wm2, sigmoid -> d_out ----
    v8f accO[4];
    gemm_from_lds<4>(my, wm2T, accO, lane);

    int n  = lane & 15;
    int ro = (lane >> 4) << 3;
#pragma unroll
    for (int t = 0; t < 4; ++t) {
        int col  = (t << 4) + n;
        float bb = bm2[col];
#pragma unroll
        for (int v = 0; v < 8; ++v) {
            long row = rowBase + ro + v;
            if (row < N) {
                float z = accO[t][v] + bb;
                out[row * 64 + col] = 1.0f / (1.0f + __expf(-z));
            }
        }
    }
}

// ------------------------------- launcher ------------------------------------

extern "C" void kernel_launch(void* const* d_in, const int* in_sizes, int n_in,
                              void* d_out, int out_size, void* d_ws, size_t ws_size,
                              hipStream_t stream) {
    (void)n_in; (void)out_size; (void)ws_size;

    const float* x    = (const float*)d_in[0];
    const int*   src  = (const int*)  d_in[1];
    const int*   dst  = (const int*)  d_in[2];
    const float* eps1 = (const float*)d_in[3];
    const float* eps2 = (const float*)d_in[4];
    const float* w11  = (const float*)d_in[5];
    const float* b11  = (const float*)d_in[6];
    const float* w12  = (const float*)d_in[7];
    const float* b12  = (const float*)d_in[8];
    const float* w21  = (const float*)d_in[9];
    const float* b21  = (const float*)d_in[10];
    const float* w22  = (const float*)d_in[11];
    const float* b22  = (const float*)d_in[12];
    const float* wm1  = (const float*)d_in[13];
    const float* bm1  = (const float*)d_in[14];
    const float* wm2  = (const float*)d_in[15];
    const float* bm2  = (const float*)d_in[16];

    const int N = in_sizes[0] / KDIM;   // nodes
    const int E = in_sizes[1];          // edges

    // Workspace layout: 6 transposed bf16 weights, then agg[N][128] f32,
    // then h1[N][128] f32.
    char* base = (char*)d_ws;
    bf16_t* w11T = (bf16_t*)(base + 0);
    bf16_t* w12T = (bf16_t*)(base + 32768);
    bf16_t* w21T = (bf16_t*)(base + 65536);
    bf16_t* w22T = (bf16_t*)(base + 98304);
    bf16_t* wm1T = (bf16_t*)(base + 131072);
    bf16_t* wm2T = (bf16_t*)(base + 163840);          // 128x64 -> 16 KB
    const size_t offAgg   = 180224;
    const size_t aggBytes = (size_t)N * KDIM * sizeof(float);
    float* agg = (float*)(base + offAgg);
    float* h1  = (float*)(base + offAgg + aggBytes);

    // 1) Weight prep: transpose + f32->bf16.
    {
        const int threads = 256;
        wprep_kernel<<<(128 * 128 + threads - 1) / threads, threads, 0, stream>>>(w11, w11T, 128, 128);
        wprep_kernel<<<(128 * 128 + threads - 1) / threads, threads, 0, stream>>>(w12, w12T, 128, 128);
        wprep_kernel<<<(128 * 128 + threads - 1) / threads, threads, 0, stream>>>(w21, w21T, 128, 128);
        wprep_kernel<<<(128 * 128 + threads - 1) / threads, threads, 0, stream>>>(w22, w22T, 128, 128);
        wprep_kernel<<<(128 * 128 + threads - 1) / threads, threads, 0, stream>>>(wm1, wm1T, 128, 128);
        wprep_kernel<<<(128 *  64 + threads - 1) / threads, threads, 0, stream>>>(wm2, wm2T, 128,  64);
    }

    const long nScatter   = (long)E * 32;
    const int  scBlocks   = (int)((nScatter + 255) / 256);
    const int  ginBlocks  = (N + 127) / 128;

    // 2) Layer 1: aggregate then fused MLP update.
    hipMemsetAsync(agg, 0, aggBytes, stream);
    scatter_kernel<<<scBlocks, 256, 0, stream>>>(x, src, dst, agg, nScatter);
    gin_update_kernel<<<ginBlocks, 256, 0, stream>>>(x, agg, eps1,
                                                     w11T, b11, w12T, b12, h1, N);

    // 3) Layer 2 + head: aggregate h1, then fused (GIN2 + head MLP + sigmoid).
    hipMemsetAsync(agg, 0, aggBytes, stream);
    scatter_kernel<<<scBlocks, 256, 0, stream>>>(h1, src, dst, agg, nScatter);
    gin2_head_kernel<<<ginBlocks, 256, 0, stream>>>(h1, agg, eps2,
                                                    w21T, b21, w22T, b22,
                                                    wm1T, bm1, wm2T, bm2,
                                                    (float*)d_out, N);
}
